// FastDecoderLayer_2997887173115
// MI455X (gfx1250) — matrix operand order repaired and checked by reading, not compiled, not run on hardware
//
#include <hip/hip_runtime.h>
#include <hip/hip_bf16.h>

#define BATCH   32
#define HEADS   16
#define DH      64
#define DMODEL  1024
#define TKV     2047
#define SLEN    1024

typedef __attribute__((ext_vector_type(16))) __bf16 v16bf;
typedef __attribute__((ext_vector_type(8)))  float  v8f;

union Frag { v16bf v; uint4 u[2]; };

__device__ __forceinline__ unsigned short f2bf(float f) {
    unsigned int u = __float_as_uint(f);
    unsigned int r = (u + 0x7FFFu + ((u >> 16) & 1u)) >> 16;   // RNE
    return (unsigned short)r;
}
__device__ __forceinline__ float bf2f(unsigned int s) {
    return __uint_as_float((s & 0xFFFFu) << 16);
}

// ---------------------------------------------------------------------------
// LayerNorm: x [B,D] -> y [B,D]
// ---------------------------------------------------------------------------
__global__ __launch_bounds__(256)
void ln_kernel(const float* __restrict__ x, const float* __restrict__ g,
               const float* __restrict__ be, float* __restrict__ y)
{
    const int b = blockIdx.x, tid = threadIdx.x;
    const float* xb = x + (size_t)b * DMODEL;
    float s = 0.f, ss = 0.f;
    for (int i = tid; i < DMODEL; i += 256) { float v = xb[i]; s += v; ss += v * v; }
    for (int off = 16; off > 0; off >>= 1) {
        s  += __shfl_xor(s,  off, 32);
        ss += __shfl_xor(ss, off, 32);
    }
    __shared__ float rs[8], rss[8];
    __shared__ float mean_s, rstd_s;
    if ((tid & 31) == 0) { rs[tid >> 5] = s; rss[tid >> 5] = ss; }
    __syncthreads();
    if (tid == 0) {
        float S = 0.f, SS = 0.f;
        for (int i = 0; i < 8; ++i) { S += rs[i]; SS += rss[i]; }
        float mean = S * (1.0f / DMODEL);
        float var  = SS * (1.0f / DMODEL) - mean * mean;
        mean_s = mean;
        rstd_s = rsqrtf(var + 1e-5f);
    }
    __syncthreads();
    const float mean = mean_s, rstd = rstd_s;
    for (int i = tid; i < DMODEL; i += 256)
        y[(size_t)b * DMODEL + i] = (xb[i] - mean) * rstd * g[i] + be[i];
}

// ---------------------------------------------------------------------------
// GEMV: out[b,n] = act(dot(Y[b,:], W[n,:]) + bias[n]) (+resid[b,n])
// one wave per output element; bandwidth-bound on W
// ---------------------------------------------------------------------------
__global__ __launch_bounds__(256)
void gemv_kernel(const float* __restrict__ Y, const float* __restrict__ W,
                 const float* __restrict__ bias, const float* __restrict__ resid,
                 float* __restrict__ out, int N, int K, int act)
{
    const int b = blockIdx.y;
    const int n = blockIdx.x * 8 + (threadIdx.x >> 5);
    const int l = threadIdx.x & 31;
    const float4* y4 = (const float4*)(Y + (size_t)b * K);
    const float4* w4 = (const float4*)(W + (size_t)n * K);
    const int K4 = K >> 2;
    float acc = 0.f;
    for (int i = l; i < K4; i += 32) {
        float4 a = y4[i], c = w4[i];
        acc += a.x * c.x + a.y * c.y + a.z * c.z + a.w * c.w;
    }
    for (int off = 16; off > 0; off >>= 1) acc += __shfl_xor(acc, off, 32);
    if (l == 0) {
        float r = acc + bias[n];
        if (act == 1) r = r * (1.0f / (1.0f + __expf(-r)));   // SiLU
        if (resid) r += resid[(size_t)b * N + n];
        out[(size_t)b * N + n] = r;
    }
}

// ---------------------------------------------------------------------------
// Self-attention flash decode over Tkv cached keys + 1 new key.
// grid (H, B), block 256 (8 waves). Lane owns dims {2l, 2l+1}.
// ---------------------------------------------------------------------------
__global__ __launch_bounds__(256)
void attn_self_kernel(const float* __restrict__ qkv,      // [B, 3D]
                      const float* __restrict__ cache_k,  // [B,H,Tkv,DH]
                      const float* __restrict__ cache_v,
                      const float* __restrict__ tgt,      // [B, D]
                      float* __restrict__ h1)             // [B, D]
{
    const int h = blockIdx.x, b = blockIdx.y;
    const int w = threadIdx.x >> 5, l = threadIdx.x & 31;
    const int T = TKV + 1;
    const float* qp = qkv + (size_t)b * (3 * DMODEL) + h * DH;
    const float q0 = qp[2 * l] * 0.125f;
    const float q1 = qp[2 * l + 1] * 0.125f;
    const float* kb = cache_k + ((size_t)b * HEADS + h) * TKV * DH;
    const float* vb = cache_v + ((size_t)b * HEADS + h) * TKV * DH;
    const float* knew = qkv + (size_t)b * (3 * DMODEL) + DMODEL + h * DH;
    const float* vnew = qkv + (size_t)b * (3 * DMODEL) + 2 * DMODEL + h * DH;

    float m = -1e30f, lsum = 0.f, o0 = 0.f, o1 = 0.f;
    for (int t = w; t < T; t += 8) {
        const float* kp = (t < TKV) ? (kb + (size_t)t * DH) : knew;
        const float* vp = (t < TKV) ? (vb + (size_t)t * DH) : vnew;
        float s = q0 * kp[2 * l] + q1 * kp[2 * l + 1];
        for (int off = 16; off > 0; off >>= 1) s += __shfl_xor(s, off, 32);
        float nm = fmaxf(m, s);
        float sc = __expf(m - nm);
        float p  = __expf(s - nm);
        o0 = o0 * sc + p * vp[2 * l];
        o1 = o1 * sc + p * vp[2 * l + 1];
        lsum = lsum * sc + p;
        m = nm;
    }
    __shared__ float sm[8], sl[8], so[8][64];
    if (l == 0) { sm[w] = m; sl[w] = lsum; }
    so[w][2 * l] = o0; so[w][2 * l + 1] = o1;
    __syncthreads();
    if (w == 0) {
        float M = -1e30f;
        for (int i = 0; i < 8; ++i) M = fmaxf(M, sm[i]);
        float L = 0.f, a0 = 0.f, a1 = 0.f;
        for (int i = 0; i < 8; ++i) {
            float e = __expf(sm[i] - M);
            L += sl[i] * e; a0 += so[i][2 * l] * e; a1 += so[i][2 * l + 1] * e;
        }
        float inv = 1.0f / L;
        size_t base = (size_t)b * DMODEL + h * DH;
        h1[base + 2 * l]     = tgt[base + 2 * l]     + a0 * inv;
        h1[base + 2 * l + 1] = tgt[base + 2 * l + 1] + a1 * inv;
    }
}

// ---------------------------------------------------------------------------
// Cross K/V projection, WMMA bf16: for each b,
//   [K|V](n) = memory[b] @ w_cross[D+n,:]^T + b_cross[D+n],  n in [0,2048)
// 128x128 tile / workgroup, BK=32, emits v_wmma_f32_16x16x32_bf16.
// Output stored bf16 in [B,H,S,DH] layout for coalesced attention reads.
// ---------------------------------------------------------------------------
#define BM 128
#define BN 128
#define BK 32
#define LDA 40   // padded row stride (ushorts): 80B, conflict-free vs 64 banks

__global__ __launch_bounds__(256)
void kv_proj_wmma_kernel(const float* __restrict__ mem,    // [B,S,K]
                         const float* __restrict__ wc,     // [3D, K]
                         const float* __restrict__ bc,     // [3D]
                         unsigned short* __restrict__ kc,  // [B,H,S,DH] bf16
                         unsigned short* __restrict__ vc)
{
    __shared__ unsigned short ldsA[BM * LDA];
    __shared__ unsigned short ldsB[BN * LDA];
    const int s0 = blockIdx.x * BM;
    const int n0 = blockIdx.y * BN;
    const int b  = blockIdx.z;
    const int tid = threadIdx.x;
    const int w = tid >> 5, l = tid & 31;
    const int lo = l & 15, hi = l >> 4;
    const int wr = w & 1;     // M half  (64 rows)
    const int wq = w >> 1;    // N quarter (32 cols)

    v8f acc[4][2] = {};

    const int lr = tid >> 3;          // loader row 0..31
    const int lc = (tid & 7) * 4;     // loader col (floats)

    for (int k0 = 0; k0 < DMODEL; k0 += BK) {
        #pragma unroll
        for (int rr = 0; rr < 4; ++rr) {
            const int row = lr + rr * 32;
            float4 a = *(const float4*)(mem + ((size_t)b * SLEN + (s0 + row)) * DMODEL + k0 + lc);
            unsigned short* da = &ldsA[row * LDA + lc];
            da[0] = f2bf(a.x); da[1] = f2bf(a.y); da[2] = f2bf(a.z); da[3] = f2bf(a.w);
            float4 wv = *(const float4*)(wc + (size_t)(DMODEL + n0 + row) * DMODEL + k0 + lc);
            unsigned short* db = &ldsB[row * LDA + lc];
            db[0] = f2bf(wv.x); db[1] = f2bf(wv.y); db[2] = f2bf(wv.z); db[3] = f2bf(wv.w);
        }
        __syncthreads();

        // B fragments: lane lo = N col, hi selects K {0-15}/{16-31}
        Frag bf[2];
        #pragma unroll
        for (int nt = 0; nt < 2; ++nt) {
            const int row = wq * 32 + nt * 16 + lo;
            bf[nt].u[0] = *(const uint4*)&ldsB[row * LDA + 16 * hi];
            bf[nt].u[1] = *(const uint4*)&ldsB[row * LDA + 16 * hi + 8];
        }
        // A fragments: lane lo = M row, hi selects K {0-15}/{16-31}
        #pragma unroll
        for (int mt = 0; mt < 4; ++mt) {
            const int row = wr * 64 + mt * 16 + lo;
            Frag af;
            af.u[0] = *(const uint4*)&ldsA[row * LDA + 16 * hi];
            af.u[1] = *(const uint4*)&ldsA[row * LDA + 16 * hi + 8];
            #pragma unroll
            for (int nt = 0; nt < 2; ++nt)
                acc[mt][nt] = __builtin_amdgcn_wmma_f32_16x16x32_bf16(
                    false, af.v, false, bf[nt].v, (short)0, acc[mt][nt], false, false);
        }
        __syncthreads();
    }

    // epilogue: add bias, pack bf16, scatter to head-major K/V
    #pragma unroll
    for (int mt = 0; mt < 4; ++mt) {
        #pragma unroll
        for (int nt = 0; nt < 2; ++nt) {
            const int nglob = n0 + wq * 32 + nt * 16 + lo;
            const float bias = bc[DMODEL + nglob];
            const int nn = nglob & (DMODEL - 1);
            const int head = nn >> 6, dh = nn & 63;
            unsigned short* dst = (nglob < DMODEL) ? kc : vc;
            #pragma unroll
            for (int r = 0; r < 8; ++r) {
                const int m = s0 + wr * 64 + mt * 16 + r + 8 * hi;
                dst[(((size_t)b * HEADS + head) * SLEN + m) * DH + dh] =
                    f2bf(acc[mt][nt][r] + bias);
            }
        }
    }
}

// ---------------------------------------------------------------------------
// Cross-attention flash decode over S=1024 bf16 K/V.
// ---------------------------------------------------------------------------
__global__ __launch_bounds__(256)
void attn_cross_kernel(const float* __restrict__ qc,           // [B, D]
                       const unsigned short* __restrict__ kcb, // [B,H,S,DH] bf16
                       const unsigned short* __restrict__ vcb,
                       const float* __restrict__ h1,
                       float* __restrict__ h2)
{
    const int h = blockIdx.x, b = blockIdx.y;
    const int w = threadIdx.x >> 5, l = threadIdx.x & 31;
    const float* qp = qc + (size_t)b * DMODEL + h * DH;
    const float q0 = qp[2 * l] * 0.125f;
    const float q1 = qp[2 * l + 1] * 0.125f;
    const unsigned int* kb = (const unsigned int*)kcb + ((size_t)b * HEADS + h) * SLEN * (DH / 2);
    const unsigned int* vb = (const unsigned int*)vcb + ((size_t)b * HEADS + h) * SLEN * (DH / 2);

    float m = -1e30f, lsum = 0.f, o0 = 0.f, o1 = 0.f;
    for (int t = w; t < SLEN; t += 8) {
        unsigned int ku = kb[(size_t)t * (DH / 2) + l];
        float s = q0 * bf2f(ku) + q1 * bf2f(ku >> 16);
        for (int off = 16; off > 0; off >>= 1) s += __shfl_xor(s, off, 32);
        float nm = fmaxf(m, s);
        float sc = __expf(m - nm);
        float p  = __expf(s - nm);
        unsigned int vu = vb[(size_t)t * (DH / 2) + l];
        o0 = o0 * sc + p * bf2f(vu);
        o1 = o1 * sc + p * bf2f(vu >> 16);
        lsum = lsum * sc + p;
        m = nm;
    }
    __shared__ float sm[8], sl[8], so[8][64];
    if (l == 0) { sm[w] = m; sl[w] = lsum; }
    so[w][2 * l] = o0; so[w][2 * l + 1] = o1;
    __syncthreads();
    if (w == 0) {
        float M = -1e30f;
        for (int i = 0; i < 8; ++i) M = fmaxf(M, sm[i]);
        float L = 0.f, a0 = 0.f, a1 = 0.f;
        for (int i = 0; i < 8; ++i) {
            float e = __expf(sm[i] - M);
            L += sl[i] * e; a0 += so[i][2 * l] * e; a1 += so[i][2 * l + 1] * e;
        }
        float inv = 1.0f / L;
        size_t base = (size_t)b * DMODEL + h * DH;
        h2[base + 2 * l]     = h1[base + 2 * l]     + a0 * inv;
        h2[base + 2 * l + 1] = h1[base + 2 * l + 1] + a1 * inv;
    }
}

// ---------------------------------------------------------------------------
extern "C" void kernel_launch(void* const* d_in, const int* in_sizes, int n_in,
                              void* d_out, int out_size, void* d_ws, size_t ws_size,
                              hipStream_t stream)
{
    const float* tgt     = (const float*)d_in[0];
    const float* memory  = (const float*)d_in[1];
    const float* cache_k = (const float*)d_in[2];
    const float* cache_v = (const float*)d_in[3];
    const float* w_self  = (const float*)d_in[4];
    const float* b_self  = (const float*)d_in[5];
    const float* w_cross = (const float*)d_in[6];
    const float* b_cross = (const float*)d_in[7];
    const float* ln1_g   = (const float*)d_in[8];
    const float* ln1_b   = (const float*)d_in[9];
    const float* ln2_g   = (const float*)d_in[10];
    const float* ln2_b   = (const float*)d_in[11];
    const float* w1      = (const float*)d_in[12];
    const float* b1      = (const float*)d_in[13];
    const float* w2      = (const float*)d_in[14];
    const float* b2      = (const float*)d_in[15];
    float* out = (float*)d_out;

    char* ws = (char*)d_ws;
    const size_t BD = (size_t)BATCH * DMODEL * sizeof(float);     // 128 KB
    float* yln  = (float*)(ws);
    float* zln  = (float*)(ws + BD);
    float* qkv  = (float*)(ws + 2 * BD);                           // [B,3D]
    float* h1   = (float*)(ws + 5 * BD);
    float* qc   = (float*)(ws + 6 * BD);
    float* h2   = (float*)(ws + 7 * BD);
    float* u    = (float*)(ws + 8 * BD);                           // [B,4D]
    unsigned short* kcb = (unsigned short*)(ws + 12 * BD);         // 64 MB bf16
    unsigned short* vcb = kcb + (size_t)BATCH * HEADS * SLEN * DH;

    // 1) y = LN1(tgt)
    hipLaunchKernelGGL(ln_kernel, dim3(BATCH), dim3(256), 0, stream, tgt, ln1_g, ln1_b, yln);
    // 2) qkv = y @ w_self^T + b_self
    hipLaunchKernelGGL(gemv_kernel, dim3(3 * DMODEL / 8, BATCH), dim3(256), 0, stream,
                       yln, w_self, b_self, (const float*)nullptr, qkv, 3 * DMODEL, DMODEL, 0);
    // 3) self-attention over cache + new kv; h1 = tgt + sa
    hipLaunchKernelGGL(attn_self_kernel, dim3(HEADS, BATCH), dim3(256), 0, stream,
                       qkv, cache_k, cache_v, tgt, h1);
    // 4) z = LN2(h1)
    hipLaunchKernelGGL(ln_kernel, dim3(BATCH), dim3(256), 0, stream, h1, ln2_g, ln2_b, zln);
    // 5) qc = z @ w_cross[:D]^T + b_cross[:D]
    hipLaunchKernelGGL(gemv_kernel, dim3(DMODEL / 8, BATCH), dim3(256), 0, stream,
                       zln, w_cross, b_cross, (const float*)nullptr, qc, DMODEL, DMODEL, 0);
    // 6) kc/vc = memory @ w_cross[D:3D]^T + bias  (bf16 WMMA, the big GEMM)
    hipLaunchKernelGGL(kv_proj_wmma_kernel, dim3(SLEN / BM, 2 * DMODEL / BN, BATCH), dim3(256), 0, stream,
                       memory, w_cross, b_cross, kcb, vcb);
    // 7) cross attention; h2 = h1 + ca
    hipLaunchKernelGGL(attn_cross_kernel, dim3(HEADS, BATCH), dim3(256), 0, stream,
                       qc, kcb, vcb, h1, h2);
    // 8) u = SiLU(h2 @ w1^T + b1)
    hipLaunchKernelGGL(gemv_kernel, dim3(4 * DMODEL / 8, BATCH), dim3(256), 0, stream,
                       h2, w1, b1, (const float*)nullptr, u, 4 * DMODEL, DMODEL, 1);
    // 9) out = h2 + u @ w2^T + b2
    hipLaunchKernelGGL(gemv_kernel, dim3(DMODEL / 8, BATCH), dim3(256), 0, stream,
                       u, w2, b2, h2, out, DMODEL, 4 * DMODEL, 0);
}
